// AssignmentLoss_57226144252719
// MI455X (gfx1250) — matrix operand order repaired and checked by reading, not compiled, not run on hardware
//
#include <hip/hip_runtime.h>
#include <hip/hip_bf16.h>
#include <math.h>

// ---------------- problem constants ----------------
#define B_    64
#define N_    1024
#define C_    558
#define CP1_  559
#define CPAD_ 576          // 36 * 16, padded column count (pad K=0, pad v=0)
#define ITERS_ 20
#define NU_   (1.0f / 559.0f)

typedef _Float16 half_t;
typedef __attribute__((ext_vector_type(8)))  half_t v8h;
typedef __attribute__((ext_vector_type(16))) half_t v16h;
typedef __attribute__((ext_vector_type(8)))  float  v8f;

// ---------------- workspace layout (bytes) ----------------
static constexpr size_t NKE     = (size_t)B_ * N_ * CPAD_;       // K elements
static constexpr size_t OFF_K   = 0;
static constexpr size_t OFF_KT  = OFF_K  + NKE * 2;              // f16 K
static constexpr size_t OFF_U   = OFF_KT + NKE * 2;              // f16 K^T
static constexpr size_t OFF_MU  = OFF_U  + (size_t)B_ * N_ * 4;
static constexpr size_t OFF_V   = OFF_MU + (size_t)B_ * N_ * 4;
static constexpr size_t OFF_LSE = OFF_V  + (size_t)B_ * CPAD_ * 4;
// total ~152 MB — fits in the MI455X 192MB L2 alongside streaming I/O

// ---------------- kernel 0: mu[b,n] = visible ? 1/max(nv,1) : 0 ----------------
__global__ __launch_bounds__(256) void k_mu(const int* __restrict__ mask,
                                            float* __restrict__ mu) {
  int b = blockIdx.x;
  __shared__ int   red[256];
  __shared__ float s_inv;
  int t = threadIdx.x;
  int cnt = 0;
  for (int n = t; n < N_; n += 256) cnt += (mask[b * N_ + n] != 0) ? 1 : 0;
  red[t] = cnt; __syncthreads();
  for (int s = 128; s > 0; s >>= 1) {
    if (t < s) red[t] += red[t + s];
    __syncthreads();
  }
  if (t == 0) s_inv = 1.0f / fmaxf((float)red[0], 1.0f);
  __syncthreads();
  float inv = s_inv;
  for (int n = t; n < N_; n += 256)
    mu[b * N_ + n] = (mask[b * N_ + n] != 0) ? inv : 0.0f;
}

// ---------------- kernel 1: v init = 1 (real cols), 0 (pad cols) ----------------
__global__ void k_initv(float* __restrict__ v) {
  int i = blockIdx.x * 256 + threadIdx.x;
  if (i < B_ * CPAD_) v[i] = ((i % CPAD_) < CP1_) ? 1.0f : 0.0f;
}

// ---------------- kernel 2: per-row softmax -> f16 K, store lse ----------------
__global__ __launch_bounds__(128) void k_prep(const float* __restrict__ logits,
                                              const float* __restrict__ dust,
                                              half_t* __restrict__ K,
                                              float* __restrict__ lse) {
  int row = blockIdx.x;                       // row = b*N + n
  const float* L = logits + (size_t)row * C_;
  __shared__ float red[128];
  int t = threadIdx.x;

  float m = -INFINITY;
  for (int c = t; c < C_; c += 128) m = fmaxf(m, L[c]);
  red[t] = m; __syncthreads();
  for (int s = 64; s > 0; s >>= 1) { if (t < s) red[t] = fmaxf(red[t], red[t + s]); __syncthreads(); }
  m = red[0]; __syncthreads();

  float sum = 0.0f;
  for (int c = t; c < C_; c += 128) sum += __expf(L[c] - m);
  red[t] = sum; __syncthreads();
  for (int s = 64; s > 0; s >>= 1) { if (t < s) red[t] += red[t + s]; __syncthreads(); }
  float l = m + __logf(red[0]);
  if (t == 0) lse[row] = l;

  float dval = __expf(dust[0]);               // dustbin score / TEMP, TEMP==1
  half_t* Kr = K + (size_t)row * CPAD_;
  for (int c = t; c < CPAD_; c += 128) {
    float kv = (c < C_) ? __expf(L[c] - l) : ((c == C_) ? dval : 0.0f);
    Kr[c] = (half_t)kv;
  }
}

// ---------------- kernel 3: LDS-tiled transpose K -> K^T ----------------
__global__ __launch_bounds__(256) void k_transpose(const half_t* __restrict__ K,
                                                   half_t* __restrict__ KT) {
  int b  = blockIdx.x;
  int c0 = blockIdx.y * 32;
  int n0 = blockIdx.z * 32;
  __shared__ half_t tile[32][33];
  int x = threadIdx.x, y = threadIdx.y;       // (32, 8)
  #pragma unroll
  for (int ii = 0; ii < 4; ++ii) {
    int r = y + ii * 8;
    tile[r][x] = K[((size_t)b * N_ + n0 + r) * CPAD_ + c0 + x];   // coalesced read
  }
  __syncthreads();
  #pragma unroll
  for (int ii = 0; ii < 4; ++ii) {
    int r = y + ii * 8;
    KT[((size_t)b * CPAD_ + c0 + r) * N_ + n0 + x] = tile[x][r];  // coalesced write
  }
}

// Build v16h A-operand per ISA 16-bit A layout from two b128 loads.
__device__ inline v16h make16(v8h lo, v8h hi) {
  return __builtin_shufflevector(lo, hi, 0,1,2,3,4,5,6,7,8,9,10,11,12,13,14,15);
}

// ---------------- kernel 4: u[n] = mu[n] / sum_c K[n,c] v[c]  (WMMA) ----------------
__global__ __launch_bounds__(256) void k_row(const half_t* __restrict__ K,
                                             const float* __restrict__ v,
                                             const float* __restrict__ mu,
                                             float* __restrict__ u) {
  int b = blockIdx.x;
  __shared__ __align__(16) half_t vh[CPAD_];
  for (int c = threadIdx.x; c < CPAD_; c += 256) vh[c] = (half_t)v[b * CPAD_ + c];
  __syncthreads();

  int wave = threadIdx.x >> 5, lane = threadIdx.x & 31;
  int hi = lane >> 4, m = lane & 15;
  int n0 = (blockIdx.y * 8 + wave) * 16;                 // 64 tiles exactly
  const half_t* Kr = K + ((size_t)b * N_ + n0 + m) * CPAD_;

  v8f acc = {};
  for (int c0 = 0; c0 < CPAD_; c0 += 32) {
    // A: 16x32 K tile (lane<16: K 0..7 & 16..23; lane>=16: 8..15 & 24..31)
    v16h a = make16(*(const v8h*)(Kr + c0 + hi * 8),
                    *(const v8h*)(Kr + c0 + 16 + hi * 8));
    // B: v broadcast across all 16 columns (lane<16: k 0..7 & 16..23; lane>=16: 8..15 & 24..31)
    v16h bb = make16(*(const v8h*)(vh + c0 + hi * 8),
                     *(const v8h*)(vh + c0 + 16 + hi * 8));
    acc = __builtin_amdgcn_wmma_f32_16x16x32_f16(false, a, false, bb,
                                                 (short)0, acc, false, false);
  }
  if (m == 0) {                       // lanes 0 (M=0..7) and 16 (M=8..15) write
    #pragma unroll
    for (int g = 0; g < 8; ++g) {
      size_t idx = (size_t)b * N_ + n0 + hi * 8 + g;
      u[idx] = mu[idx] / acc[g];
    }
  }
}

// ---------------- kernel 5: v[c] = nu / sum_n K[n,c] u[n]  (WMMA on K^T) ----------------
__global__ __launch_bounds__(192) void k_col(const half_t* __restrict__ KT,
                                             const float* __restrict__ u,
                                             float* __restrict__ v) {
  int b = blockIdx.x;
  __shared__ __align__(16) half_t uh[N_];
  for (int n = threadIdx.x; n < N_; n += 192) uh[n] = (half_t)u[b * N_ + n];
  __syncthreads();

  int wave = threadIdx.x >> 5, lane = threadIdx.x & 31;
  int hi = lane >> 4, m = lane & 15;
  int c0w = (blockIdx.y * 6 + wave) * 16;                // 36 tiles exactly
  const half_t* Kr = KT + ((size_t)b * CPAD_ + c0w + m) * N_;

  v8f acc = {};
  for (int n0 = 0; n0 < N_; n0 += 32) {
    v16h a = make16(*(const v8h*)(Kr + n0 + hi * 8),
                    *(const v8h*)(Kr + n0 + 16 + hi * 8));
    v16h bb = make16(*(const v8h*)(uh + n0 + hi * 8),
                     *(const v8h*)(uh + n0 + 16 + hi * 8));
    acc = __builtin_amdgcn_wmma_f32_16x16x32_f16(false, a, false, bb,
                                                 (short)0, acc, false, false);
  }
  if (m == 0) {
    #pragma unroll
    for (int g = 0; g < 8; ++g) {
      int c = c0w + hi * 8 + g;
      v[(size_t)b * CPAD_ + c] = (c < CP1_) ? (NU_ / acc[g]) : 0.0f;
    }
  }
}

// ---------------- kernel 6: P = exp(logits - lse) * u * v (exact K) ----------------
__global__ __launch_bounds__(128) void k_final(const float* __restrict__ logits,
                                               const float* __restrict__ dust,
                                               const float* __restrict__ lse,
                                               const float* __restrict__ u,
                                               const float* __restrict__ v,
                                               float* __restrict__ out) {
  int row = blockIdx.x;               // b*N + n
  int b = row >> 10;
  float l = lse[row];
  float ur = u[row];
  float dval = __expf(dust[0]);
  const float* L = logits + (size_t)row * C_;
  float* O = out + (size_t)row * CP1_;
  for (int c = threadIdx.x; c < CP1_; c += 128) {
    float kv = (c < C_) ? __expf(L[c] - l) : dval;
    O[c] = kv * ur * v[b * CPAD_ + c];
  }
}

// ---------------- host launcher ----------------
extern "C" void kernel_launch(void* const* d_in, const int* in_sizes, int n_in,
                              void* d_out, int out_size, void* d_ws, size_t ws_size,
                              hipStream_t stream) {
  const float* logits = (const float*)d_in[0];   // [B,N,C] f32
  const int*   mask   = (const int*)d_in[1];     // [B,N] bool->int
  const float* dust   = (const float*)d_in[2];   // [1] f32

  char* ws = (char*)d_ws;
  half_t* K   = (half_t*)(ws + OFF_K);
  half_t* KT  = (half_t*)(ws + OFF_KT);
  float*  u   = (float*) (ws + OFF_U);
  float*  mu  = (float*) (ws + OFF_MU);
  float*  v   = (float*) (ws + OFF_V);
  float*  lse = (float*) (ws + OFF_LSE);

  k_mu<<<B_, 256, 0, stream>>>(mask, mu);
  k_initv<<<(B_ * CPAD_ + 255) / 256, 256, 0, stream>>>(v);
  k_prep<<<B_ * N_, 128, 0, stream>>>(logits, dust, K, lse);
  k_transpose<<<dim3(B_, CPAD_ / 32, N_ / 32), dim3(32, 8), 0, stream>>>(K, KT);

  for (int it = 0; it < ITERS_; ++it) {
    k_row<<<dim3(B_, 8), 256, 0, stream>>>(K, v, mu, u);
    k_col<<<dim3(B_, 6), 192, 0, stream>>>(KT, u, v);
  }

  k_final<<<B_ * N_, 128, 0, stream>>>(logits, dust, lse, u, v, (float*)d_out);
}
